// LDS_63720134804240
// MI455X (gfx1250) — matrix operand order repaired and checked by reading, not compiled, not run on hardware
//
#include <hip/hip_runtime.h>
#include <hip/hip_bf16.h>

typedef __attribute__((ext_vector_type(16))) _Float16 v16h;
typedef __attribute__((ext_vector_type(8)))  _Float16 v8h;
typedef __attribute__((ext_vector_type(2)))  _Float16 v2h;
typedef __attribute__((ext_vector_type(8)))  float    v8f;

#define BB  8
#define TT  4096
#define DD  128
#define NN  1024
#define OO  128
#define KXC 10
#define BT  (BB*TT)   // 32768 rows

// ---------------- pack kernels: f32 -> f16, weights transposed to [col][k] ----

__global__ void k_pack_x(const float* __restrict__ x, _Float16* __restrict__ xh, int n) {
    int i = blockIdx.x * blockDim.x + threadIdx.x;
    if (i < n) xh[i] = (_Float16)x[i];
}

__global__ void k_pack_B(const float* __restrict__ Bm, _Float16* __restrict__ Bp) {
    int i = blockIdx.x * blockDim.x + threadIdx.x;           // over N*D
    if (i < NN * DD) {
        int n = i / DD, d = i % DD;
        Bp[i] = (_Float16)Bm[d * NN + n];                    // Bp[n][d]
    }
}

__global__ void k_pack_C(const float* __restrict__ C, _Float16* __restrict__ Cp) {
    int i = blockIdx.x * blockDim.x + threadIdx.x;           // over O*N
    if (i < OO * NN) {
        int o = i / NN, n = i % NN;
        Cp[i] = (_Float16)C[n * OO + o];                     // Cp[o][n]
    }
}

__global__ void k_pack_M(const float* __restrict__ M, _Float16* __restrict__ Mp) {
    int i = blockIdx.x * blockDim.x + threadIdx.x;           // over KX*O*D
    if (i < KXC * OO * DD) {
        int kx = i / (OO * DD);
        int r  = i % (OO * DD);
        int o  = r / DD, d = r % DD;
        Mp[i] = (_Float16)M[(o * DD + d) * KXC + kx];        // Mp[kx][o][d]
    }
}

// ---------------- WMMA fragment loaders (wave32 layouts per CDNA5 ISA) -------

// A 16x32 f16: lane L -> row L&15; lanes 0-15 K:{0..7,16..23}, lanes 16-31 K:{8..15,24..31}
__device__ __forceinline__ v16h load_a_frag(const _Float16* __restrict__ base,
                                            int stride, int lane) {
    int row = lane & 15;
    int kb  = (lane & 16) ? 8 : 0;
    const _Float16* p = base + (size_t)row * stride + kb;
    v8h lo = *(const v8h*)(p);
    v8h hi = *(const v8h*)(p + 16);
    v16h a;
#pragma unroll
    for (int j = 0; j < 8; ++j) { a[j] = lo[j]; a[j + 8] = hi[j]; }
    return a;
}

// B 32x16 f16, operand pre-transposed to [col][k]:
// lane L -> col L&15; lanes 0-15 K:{0..7,16..23}, lanes 16-31 K:{8..15,24..31}
__device__ __forceinline__ v16h load_b_frag(const _Float16* __restrict__ base,
                                            int stride, int lane) {
    int col = lane & 15;
    int kb  = (lane & 16) ? 8 : 0;
    const _Float16* p = base + (size_t)col * stride + kb;
    v8h lo = *(const v8h*)(p);
    v8h hi = *(const v8h*)(p + 16);
    v16h b;
#pragma unroll
    for (int j = 0; j < 8; ++j) { b[j] = lo[j]; b[j + 8] = hi[j]; }
    return b;
}

// ---------------- kernel 1: uB = x @ B  (f16 in, f32 accum, f16 out) ---------

__global__ void __launch_bounds__(256)
k_gemm_uB(const _Float16* __restrict__ xh, const _Float16* __restrict__ Bp,
          _Float16* __restrict__ U) {
    const int lane    = threadIdx.x & 31;
    const int wave    = threadIdx.x >> 5;
    const int rowTile = blockIdx.x >> 3;                      // 0..2047 (bt/16)
    const int colTile = ((blockIdx.x & 7) << 3) + wave;       // 0..63   (n/16)
    const int r0 = rowTile * 16;
    const int c0 = colTile * 16;

    v8f acc = {};
#pragma unroll
    for (int k = 0; k < DD; k += 32) {
        v16h a = load_a_frag(xh + (size_t)r0 * DD + k, DD, lane);
        v16h b = load_b_frag(Bp + (size_t)c0 * DD + k, DD, lane);
        acc = __builtin_amdgcn_wmma_f32_16x16x32_f16(false, a, false, b,
                                                     (short)0, acc, false, false);
    }
    const int col   = c0 + (lane & 15);
    const int rbase = r0 + ((lane & 16) ? 8 : 0);
#pragma unroll
    for (int r = 0; r < 8; ++r)
        U[(size_t)(rbase + r) * NN + col] = (_Float16)acc[r];
}

// ---------------- kernel 2: diagonal scan, in place over U -------------------
// 8192 independent chains; each thread owns 2 adjacent n (32-bit coalesced RMW).

__global__ void __launch_bounds__(256)
k_scan(_Float16* __restrict__ U, const float* __restrict__ h0,
       const float* __restrict__ A) {
    const int tid = blockIdx.x * blockDim.x + threadIdx.x;    // 0..4095
    const int b   = tid >> 9;                                 // N/2 = 512 pairs/batch
    const int n0  = (tid & 511) << 1;
    const float a0 = A[n0], a1 = A[n0 + 1];
    float ha = h0[n0], hb = h0[n0 + 1];
    _Float16* p = U + (size_t)b * TT * NN + n0;
    for (int t = 0; t < TT; ++t) {
        v2h u = *(const v2h*)p;
        ha = fmaf(a0, ha, (float)u[0]);
        hb = fmaf(a1, hb, (float)u[1]);
        v2h o; o[0] = (_Float16)ha; o[1] = (_Float16)hb;
        *(v2h*)p = o;
        p += NN;
    }
}

// ---------------- kernel 3: out = h_seq @ C + sum_i shift(x,i) @ M_i ---------

__global__ void __launch_bounds__(256)
k_out(const _Float16* __restrict__ U, const _Float16* __restrict__ Cp,
      const _Float16* __restrict__ xh, const _Float16* __restrict__ Mp,
      float* __restrict__ out) {
    const int lane = threadIdx.x & 31;
    const int wave = threadIdx.x >> 5;
    const int r0   = blockIdx.x * 16;                         // bt row base
    const int c0   = wave * 16;                               // o col base (8 waves = O)

    v8f acc = {};

    // lds_out: K over N = 1024 -> 32 WMMAs
#pragma unroll 4
    for (int k = 0; k < NN; k += 32) {
        v16h a = load_a_frag(U + (size_t)r0 * NN + k, NN, lane);
        v16h b = load_b_frag(Cp + (size_t)c0 * NN + k, NN, lane);
        acc = __builtin_amdgcn_wmma_f32_16x16x32_f16(false, a, false, b,
                                                     (short)0, acc, false, false);
    }

    // AR: 10 time-shifted GEMMs, K over D = 128 -> 40 WMMAs
    const int arow  = lane & 15;
    const int kbA   = (lane & 16) ? 8 : 0;
    const int myRow = r0 + arow;                              // global bt row of lane's A row
    const int tloc  = myRow & (TT - 1);                       // time within batch
#pragma unroll
    for (int i = 0; i < KXC; ++i) {
        const bool valid = (tloc - i) >= 0;                   // causal zero padding
        const _Float16* rp = xh + (size_t)(myRow - i) * DD;
#pragma unroll
        for (int k = 0; k < DD; k += 32) {
            v16h a = {};
            if (valid) {
                const _Float16* p = rp + k + kbA;
                v8h lo = *(const v8h*)(p);
                v8h hi = *(const v8h*)(p + 16);
#pragma unroll
                for (int j = 0; j < 8; ++j) { a[j] = lo[j]; a[j + 8] = hi[j]; }
            }
            v16h b = load_b_frag(Mp + ((size_t)i * OO + c0) * DD + k, DD, lane);
            acc = __builtin_amdgcn_wmma_f32_16x16x32_f16(false, a, false, b,
                                                         (short)0, acc, false, false);
        }
    }

    const int col   = c0 + (lane & 15);
    const int rbase = r0 + ((lane & 16) ? 8 : 0);
#pragma unroll
    for (int r = 0; r < 8; ++r)
        out[(size_t)(rbase + r) * OO + col] = acc[r];
}

// ---------------- host launcher ---------------------------------------------

extern "C" void kernel_launch(void* const* d_in, const int* in_sizes, int n_in,
                              void* d_out, int out_size, void* d_ws, size_t ws_size,
                              hipStream_t stream) {
    const float* inputs = (const float*)d_in[0];  // (8,4096,128)
    const float* h0     = (const float*)d_in[1];  // (1024,)
    const float* A      = (const float*)d_in[2];  // (1024,)
    const float* Bm     = (const float*)d_in[3];  // (128,1024)
    const float* C      = (const float*)d_in[4];  // (1024,128)
    const float* M      = (const float*)d_in[5];  // (128,128,10)
    float* out = (float*)d_out;

    char* ws = (char*)d_ws;
    _Float16* U  = (_Float16*)ws;                                  // BT*NN f16 = 64 MB
    size_t off   = (size_t)BT * NN * sizeof(_Float16);
    _Float16* xh = (_Float16*)(ws + off);                          // BT*DD f16 = 8 MB
    off += (size_t)BT * DD * sizeof(_Float16);
    _Float16* Bp = (_Float16*)(ws + off);                          // NN*DD f16
    off += (size_t)NN * DD * sizeof(_Float16);
    _Float16* Cp = (_Float16*)(ws + off);                          // OO*NN f16
    off += (size_t)OO * NN * sizeof(_Float16);
    _Float16* Mp = (_Float16*)(ws + off);                          // KX*OO*DD f16

    const int nx = BT * DD;
    k_pack_x<<<(nx + 255) / 256, 256, 0, stream>>>(inputs, xh, nx);
    k_pack_B<<<(NN * DD + 255) / 256, 256, 0, stream>>>(Bm, Bp);
    k_pack_C<<<(OO * NN + 255) / 256, 256, 0, stream>>>(C, Cp);
    k_pack_M<<<(KXC * OO * DD + 255) / 256, 256, 0, stream>>>(M, Mp);

    // uB GEMM: 2048 row tiles x 64 col tiles, 8 waves/block
    k_gemm_uB<<<(BT / 16) * 8, 256, 0, stream>>>(xh, Bp, U);

    // scan: B*N/2 = 4096 threads
    k_scan<<<(BB * NN / 2) / 256, 256, 0, stream>>>(U, h0, A);

    // output GEMM + AR: one block per 16-row tile, 8 waves cover O=128
    k_out<<<BT / 16, 256, 0, stream>>>(U, Cp, xh, Mp, out);
}
